// CrystalAE_27599459844211
// MI455X (gfx1250) — matrix operand order repaired and checked by reading, not compiled, not run on hardware
//
#include <hip/hip_runtime.h>
#include <hip/hip_bf16.h>

typedef __attribute__((ext_vector_type(16))) _Float16 v16h;
typedef __attribute__((ext_vector_type(8)))  _Float16 v8h;
typedef __attribute__((ext_vector_type(8)))  float    v8f;

#define AF    64                 // atom feature dim A
#define NBRF  41
#define ORIG  92
#define NM    12                 // neighbors per atom
#define NN    50000              // atoms
#define BB    1000               // crystals
#define NPC   50
#define EDGES (NN*NM)            // 600000
#define KIN   (2*AF + NBRF)      // 169
#define KPAD  192                // conv K padded (6 k-steps of 32)
#define COUT  (2*AF)             // 128
#define EPAD  96                 // embed K padded (3 k-steps)
#define CPAD  96                 // atom_feat col pad (6 col tiles)
#define EPSV  1e-5f

__device__ __forceinline__ float softplus_(float x) {
    return (x > 20.f) ? x : log1pf(__expf(x));
}
__device__ __forceinline__ float sigmoid_(float x) {
    return 1.f / (1.f + __expf(-x));
}

// ---------------------------------------------------------------------------
// CDNA5 async global->LDS staging (ASYNCcnt-tracked), then wait + barrier.
// Sizes must be multiples of 16 bytes.
// ---------------------------------------------------------------------------
__device__ __forceinline__ void stage_to_lds(_Float16* lds, const _Float16* g,
                                             int halves) {
    const int bytes = halves * 2;
    const unsigned base = (unsigned)(unsigned long long)(void*)lds;
    const char* gp = (const char*)g;
    for (int off = threadIdx.x * 16; off < bytes; off += blockDim.x * 16) {
        asm volatile("global_load_async_to_lds_b128 %0, %1, off"
                     :: "v"(base + off), "v"(gp + off) : "memory");
    }
    asm volatile("s_wait_asynccnt 0x0" ::: "memory");
    __syncthreads();
}

// ---------------------------------------------------------------------------
// Weight prep: transpose, convert to f16, zero-pad.
//   wT [l][c=0..127][k=0..191] = fcw[l][k][c]
//   ebT[c=0..63 ][k=0..95 ]    = emb_w[k][c]
//   afT[c=0..95 ][k=0..63 ]    = af_w [k][c]   (c >= 92 -> 0)
// ---------------------------------------------------------------------------
__global__ void k_prep(const float* __restrict__ fcw,
                       const float* __restrict__ emb_w,
                       const float* __restrict__ af_w,
                       _Float16* __restrict__ wT,
                       _Float16* __restrict__ ebT,
                       _Float16* __restrict__ afT) {
    const int idx = blockIdx.x * blockDim.x + threadIdx.x;
    const int t1 = 3 * COUT * KPAD;            // 73728
    const int t2 = t1 + AF * EPAD;             // + 6144
    const int t3 = t2 + CPAD * AF;             // + 6144
    if (idx < t1) {
        const int l = idx / (COUT * KPAD);
        const int r = idx - l * (COUT * KPAD);
        const int c = r / KPAD, k = r - c * KPAD;
        wT[idx] = (_Float16)((k < KIN) ? fcw[(l * KIN + k) * COUT + c] : 0.f);
    } else if (idx < t2) {
        const int r = idx - t1;
        const int c = r / EPAD, k = r - c * EPAD;
        ebT[r] = (_Float16)((k < ORIG) ? emb_w[k * AF + c] : 0.f);
    } else if (idx < t3) {
        const int r = idx - t2;
        const int c = r / AF, k = r - c * AF;
        afT[r] = (_Float16)((c < ORIG) ? af_w[k * ORIG + c] : 0.f);
    }
}

// ---------------------------------------------------------------------------
// x = atom_fea @ emb_w   (50000x92)@(92x64). Weights staged in LDS (f16).
// Writes f32 x (for residual) AND f16 shadow xh (for WMMA gathers).
// ---------------------------------------------------------------------------
__global__ void k_embed(const float* __restrict__ atom_fea,
                        const _Float16* __restrict__ ebT,
                        float* __restrict__ x,
                        _Float16* __restrict__ xh) {
    __shared__ __align__(32) _Float16 ldsE[AF * EPAD];      // 12 KB
    stage_to_lds(ldsE, ebT, AF * EPAD);

    const int lane = threadIdx.x & 31, wave = threadIdx.x >> 5;
    const int tile = blockIdx.x * 8 + wave;
    if (tile >= NN / 16) return;                             // 3125, exact
    const int half = lane >> 4, lrow = lane & 15;
    const float* arow = atom_fea + (tile * 16 + lrow) * ORIG;

    v8f acc[4] = {};
#pragma unroll
    for (int ks = 0; ks < 3; ++ks) {
        const int k0 = ks * 32;
        v16h a;
#pragma unroll
        for (int g = 0; g < 2; ++g) {
            const int s = k0 + half * 16 + g * 8;
            if (s + 8 <= ORIG) {
                const float4 p = *(const float4*)(arow + s);
                const float4 q = *(const float4*)(arow + s + 4);
                a[8*g+0]=(_Float16)p.x; a[8*g+1]=(_Float16)p.y;
                a[8*g+2]=(_Float16)p.z; a[8*g+3]=(_Float16)p.w;
                a[8*g+4]=(_Float16)q.x; a[8*g+5]=(_Float16)q.y;
                a[8*g+6]=(_Float16)q.z; a[8*g+7]=(_Float16)q.w;
            } else {
#pragma unroll
                for (int e = 0; e < 8; ++e)
                    a[8*g+e] = (_Float16)((s + e < ORIG) ? arow[s + e] : 0.f);
            }
        }
#pragma unroll
        for (int t = 0; t < 4; ++t) {
            const v16h b = *(const v16h*)(ldsE + (t*16 + lrow) * EPAD + k0 + half*16);
            acc[t] = __builtin_amdgcn_wmma_f32_16x16x32_f16(
                false, a, false, b, (short)0, acc[t], false, false);
        }
    }
#pragma unroll
    for (int t = 0; t < 4; ++t) {
        const int col = t * 16 + lrow;
#pragma unroll
        for (int r = 0; r < 8; ++r) {
            const int row = tile * 16 + r + 8 * half;
            x [row * AF + col] = acc[t][r];
            xh[row * AF + col] = (_Float16)acc[t][r];
        }
    }
}

// ---------------------------------------------------------------------------
// Conv GEMM core: 16-edge strip x 128 cols, K=169 padded to 192.
// A: x-regions load pre-converted f16 (16B vector loads); nbr_fea converts.
// B: fragments from LDS (f16, transposed).
// ---------------------------------------------------------------------------
__device__ __forceinline__ v8h load_run8h(const _Float16* xi, const _Float16* xn,
                                          const float* nf, int s) {
    if (s + 8 <= AF)   return *(const v8h*)(xi + s);
    if (s + 8 <= 2*AF) return *(const v8h*)(xn + s - AF);
    v8h r;
#pragma unroll
    for (int e = 0; e < 8; ++e) {
        const int k = s + e;
        r[e] = (_Float16)((k < KIN) ? nf[k - 2*AF] : 0.f);
    }
    return r;
}

__device__ __forceinline__ void conv_gemm(const _Float16* __restrict__ xh,
                                          const float* __restrict__ nbr_fea,
                                          const int*   __restrict__ nbr_idx,
                                          const _Float16* __restrict__ ldsW,
                                          const float* __restrict__ bias,
                                          int tile, int lane, v8f acc[8]) {
    const int half = lane >> 4, lrow = lane & 15;
    const int edge = tile * 16 + lrow;
    const int i = edge / NM;
    const int nb = nbr_idx[edge];
    const _Float16* xi = xh + i * AF;
    const _Float16* xn = xh + nb * AF;
    const float* nf = nbr_fea + edge * NBRF;

#pragma unroll
    for (int ks = 0; ks < 6; ++ks) {
        const int k0 = ks * 32;
        const v8h r0 = load_run8h(xi, xn, nf, k0 + half * 16);
        const v8h r1 = load_run8h(xi, xn, nf, k0 + half * 16 + 8);
        v16h a;
#pragma unroll
        for (int e = 0; e < 8; ++e) { a[e] = r0[e]; a[8 + e] = r1[e]; }
#pragma unroll
        for (int t = 0; t < 8; ++t) {
            const v16h b = *(const v16h*)(ldsW + (t*16 + lrow) * KPAD + k0 + half*16);
            acc[t] = __builtin_amdgcn_wmma_f32_16x16x32_f16(
                false, a, false, b, (short)0, acc[t], false, false);
        }
    }
#pragma unroll
    for (int t = 0; t < 8; ++t) {
        const float bv = bias[t*16 + lrow];
#pragma unroll
        for (int r = 0; r < 8; ++r) acc[t][r] += bv;
    }
}

// Pass 1: per-channel sum / sumsq for batchnorm over all 600000 rows.
__global__ void k_conv_stats(const _Float16* __restrict__ xh,
                             const float* __restrict__ nbr_fea,
                             const int*   __restrict__ nbr_idx,
                             const _Float16* __restrict__ wT,
                             const float* __restrict__ bias,
                             float* __restrict__ stats1) {
    __shared__ __align__(32) _Float16 ldsW[COUT * KPAD];     // 48 KB
    __shared__ float lsum[COUT], lsq[COUT];
    const int tid = threadIdx.x;
    if (tid < COUT) { lsum[tid] = 0.f; lsq[tid] = 0.f; }
    stage_to_lds(ldsW, wT, COUT * KPAD);                     // includes barrier

    const int lane = tid & 31, wave = tid >> 5;
    const int tile = blockIdx.x * 8 + wave;
    if (tile < EDGES / 16) {
        v8f acc[8] = {};
        conv_gemm(xh, nbr_fea, nbr_idx, ldsW, bias, tile, lane, acc);
        const int lrow = lane & 15;
#pragma unroll
        for (int t = 0; t < 8; ++t) {
            float s = 0.f, q = 0.f;
#pragma unroll
            for (int r = 0; r < 8; ++r) { const float v = acc[t][r]; s += v; q += v*v; }
            atomicAdd(&lsum[t*16 + lrow], s);
            atomicAdd(&lsq [t*16 + lrow], q);
        }
    }
    __syncthreads();
    if (tid < COUT) {
        atomicAdd(&stats1[tid], lsum[tid]);
        atomicAdd(&stats1[COUT + tid], lsq[tid]);
    }
}

// Pass 2: recompute GEMM, fuse BN-affine + sigmoid*softplus + neighbor-sum.
__global__ void k_conv_apply(const _Float16* __restrict__ xh,
                             const float* __restrict__ nbr_fea,
                             const int*   __restrict__ nbr_idx,
                             const _Float16* __restrict__ wT,
                             const float* __restrict__ bias,
                             const float* __restrict__ stats1,
                             const float* __restrict__ g1,
                             const float* __restrict__ b1,
                             float* __restrict__ summed) {
    __shared__ __align__(32) _Float16 ldsW[COUT * KPAD];     // 48 KB
    stage_to_lds(ldsW, wT, COUT * KPAD);

    const int lane = threadIdx.x & 31, wave = threadIdx.x >> 5;
    const int tile = blockIdx.x * 8 + wave;
    if (tile >= EDGES / 16) return;

    v8f acc[8] = {};
    conv_gemm(xh, nbr_fea, nbr_idx, ldsW, bias, tile, lane, acc);

    const int half = lane >> 4, lrow = lane & 15;
    const float invE = 1.f / (float)EDGES;
    float scl[8], sft[8];
#pragma unroll
    for (int t = 0; t < 8; ++t) {
        const int c = t * 16 + lrow;
        const float m = stats1[c] * invE;
        const float v = stats1[COUT + c] * invE - m * m;
        const float s = rsqrtf(v + EPSV) * g1[c];
        scl[t] = s; sft[t] = b1[c] - m * s;
    }
#pragma unroll
    for (int t = 0; t < 4; ++t) {
        const int col = t * 16 + lrow;
#pragma unroll
        for (int r = 0; r < 8; ++r) {
            const float f = acc[t][r]     * scl[t]   + sft[t];
            const float c = acc[t + 4][r] * scl[t+4] + sft[t+4];
            const float val = sigmoid_(f) * softplus_(c);
            const int edge = tile * 16 + r + 8 * half;
            atomicAdd(&summed[(edge / NM) * AF + col], val);
        }
    }
}

// Batchnorm-2 stats over 50000 rows x 64 channels.
__global__ void k_bn2_stats(const float* __restrict__ summed,
                            float* __restrict__ stats2) {
    __shared__ float ls[256], lq[256];
    const int tid = threadIdx.x;
    const int col = tid & 63, grp = tid >> 6;
    float s = 0.f, q = 0.f;
    for (int r = blockIdx.x * 4 + grp; r < NN; r += gridDim.x * 4) {
        const float v = summed[r * AF + col];
        s += v; q += v * v;
    }
    ls[tid] = s; lq[tid] = q;
    __syncthreads();
    if (tid < 64) {
        s = ls[tid] + ls[tid+64] + ls[tid+128] + ls[tid+192];
        q = lq[tid] + lq[tid+64] + lq[tid+128] + lq[tid+192];
        atomicAdd(&stats2[tid], s);
        atomicAdd(&stats2[64 + tid], q);
    }
}

// x_out = softplus(x_in + BN2(summed));  also writes f16 shadow for next layer.
__global__ void k_bn2_apply(const float* __restrict__ xin,
                            const float* __restrict__ summed,
                            const float* __restrict__ stats2,
                            const float* __restrict__ g2,
                            const float* __restrict__ b2,
                            float* __restrict__ xout,
                            _Float16* __restrict__ xh) {
    const float invN = 1.f / (float)NN;
    for (int idx = blockIdx.x * blockDim.x + threadIdx.x; idx < NN * AF;
         idx += gridDim.x * blockDim.x) {
        const int c = idx & 63;
        const float m = stats2[c] * invN;
        const float v = stats2[64 + c] * invN - m * m;
        const float sn = (summed[idx] - m) * rsqrtf(v + EPSV) * g2[c] + b2[c];
        const float o = softplus_(xin[idx] + sn);
        xout[idx] = o;
        xh[idx] = (_Float16)o;
    }
}

// Quadratic forms + 6x6 FC + log-softmax. 32 atoms per block, LDS-staged.
__global__ void k_pool(const float* __restrict__ xf,
                       const int*   __restrict__ cidx,
                       const float* __restrict__ adj_w,
                       const float* __restrict__ adj_b,
                       const float* __restrict__ fc1_w,
                       const float* __restrict__ fc1_b,
                       float* __restrict__ logp) {
    __shared__ float v[32][AF];
    __shared__ float qr[32][6];
    const int tid = threadIdx.x;                 // blockDim = 192 (32 atoms x 6)
    const int base = blockIdx.x * 32;
    for (int t = tid; t < 32 * AF; t += 192) {
        const int s = t / AF, d = t - s * AF;
        const int g = base + s;
        v[s][d] = (g < BB * NPC) ? xf[cidx[g] * AF + d] : 0.f;
    }
    __syncthreads();
    const int s = tid / 6, o = tid - s * 6;
    if (base + s < BB * NPC) {
        const float* wo = adj_w + o * AF * AF;
        float q = 0.f;
        for (int d = 0; d < AF; ++d) {
            float inner = 0.f;
#pragma unroll 8
            for (int e = 0; e < AF; ++e) inner += wo[d * AF + e] * v[s][e];
            q += v[s][d] * inner;
        }
        qr[s][o] = q + adj_b[o];
    }
    __syncthreads();
    if (tid < 32 && (base + tid) < BB * NPC) {
        float r[6], mx = -1e30f;
#pragma unroll
        for (int oo = 0; oo < 6; ++oo) {
            float a = fc1_b[oo];
#pragma unroll
            for (int p = 0; p < 6; ++p) a += qr[tid][p] * fc1_w[p * 6 + oo];
            r[oo] = a; mx = fmaxf(mx, a);
        }
        float sum = 0.f;
#pragma unroll
        for (int oo = 0; oo < 6; ++oo) sum += __expf(r[oo] - mx);
        const float lse = mx + __logf(sum);
#pragma unroll
        for (int oo = 0; oo < 6; ++oo) logp[(base + tid) * 6 + oo] = r[oo] - lse;
    }
}

// edge_prob broadcast: out[b, n1*NPC+n2, o] = logp[b, n2, o]
__global__ void k_edge_prob(const float* __restrict__ logp,
                            float* __restrict__ out) {
    const long total = (long)BB * NPC * NPC * 6;
    for (long idx = blockIdx.x * (long)blockDim.x + threadIdx.x; idx < total;
         idx += (long)gridDim.x * blockDim.x) {
        const int o  = (int)(idx % 6);
        const long r = idx / 6;
        const int n2 = (int)(r % NPC);
        const long r2 = r / NPC;                 // b*NPC + n1
        const int b  = (int)(r2 / NPC);
        out[idx] = logp[(b * NPC + n2) * 6 + o];
    }
}

// atom_feature = cf @ af_w + af_b   (50000x64)@(64x92), cols padded to 96.
// Reads the f16 shadow directly (no per-element cvt).
__global__ void k_atom_feat(const _Float16* __restrict__ xh,
                            const int*   __restrict__ cidx,
                            const _Float16* __restrict__ afT,
                            const float* __restrict__ af_b,
                            float* __restrict__ out) {
    __shared__ __align__(32) _Float16 ldsA[CPAD * AF];       // 12 KB
    stage_to_lds(ldsA, afT, CPAD * AF);

    const int lane = threadIdx.x & 31, wave = threadIdx.x >> 5;
    const int tile = blockIdx.x * 8 + wave;
    if (tile >= (BB * NPC) / 16) return;          // 3125, exact
    const int half = lane >> 4, lrow = lane & 15;
    const _Float16* xrow = xh + cidx[tile * 16 + lrow] * AF;

    v8f acc[6] = {};
#pragma unroll
    for (int ks = 0; ks < 2; ++ks) {
        const int k0 = ks * 32;
        const v8h r0 = *(const v8h*)(xrow + k0 + half * 16);
        const v8h r1 = *(const v8h*)(xrow + k0 + half * 16 + 8);
        v16h a;
#pragma unroll
        for (int e = 0; e < 8; ++e) { a[e] = r0[e]; a[8 + e] = r1[e]; }
#pragma unroll
        for (int t = 0; t < 6; ++t) {
            const v16h b = *(const v16h*)(ldsA + (t*16 + lrow) * AF + k0 + half*16);
            acc[t] = __builtin_amdgcn_wmma_f32_16x16x32_f16(
                false, a, false, b, (short)0, acc[t], false, false);
        }
    }
#pragma unroll
    for (int t = 0; t < 6; ++t) {
        const int col = t * 16 + lrow;
        if (col < ORIG) {
            const float bv = af_b[col];
#pragma unroll
            for (int r = 0; r < 8; ++r)
                out[(tile * 16 + r + 8 * half) * ORIG + col] = acc[t][r] + bv;
        }
    }
}

extern "C" void kernel_launch(void* const* d_in, const int* in_sizes, int n_in,
                              void* d_out, int out_size, void* d_ws, size_t ws_size,
                              hipStream_t stream) {
    const float* atom_fea = (const float*)d_in[0];
    const float* nbr_fea  = (const float*)d_in[1];
    const int*   nbr_idx  = (const int*)  d_in[2];
    const int*   cidx     = (const int*)  d_in[3];
    const float* emb_w    = (const float*)d_in[4];
    const float* fcw      = (const float*)d_in[5];
    const float* fcb      = (const float*)d_in[6];
    const float* bn1g     = (const float*)d_in[7];
    const float* bn1b     = (const float*)d_in[8];
    const float* bn2g     = (const float*)d_in[9];
    const float* bn2b     = (const float*)d_in[10];
    const float* adj_w    = (const float*)d_in[11];
    const float* adj_b    = (const float*)d_in[12];
    const float* fc1_w    = (const float*)d_in[13];
    const float* fc1_b    = (const float*)d_in[14];
    const float* af_w     = (const float*)d_in[15];
    const float* af_b     = (const float*)d_in[16];
    float* out = (float*)d_out;

    // Workspace layout (~47 MB)
    float* xA     = (float*)d_ws;            // 50000*64 f32
    float* xB     = xA + NN * AF;            // 50000*64 f32
    float* summed = xB + NN * AF;            // 50000*64 f32
    float* stats1 = summed + NN * AF;        // 2*128 (sum, sumsq)
    float* stats2 = stats1 + 2 * COUT;       // 2*64
    float* logp   = stats2 + 2 * AF;         // 1000*50*6
    _Float16* wT  = (_Float16*)(logp + BB * NPC * 6);  // 3*128*192 (16B aligned)
    _Float16* ebT = wT + 3 * COUT * KPAD;              // 64*96
    _Float16* afT = ebT + AF * EPAD;                   // 96*64
    _Float16* xh  = afT + CPAD * AF;                   // 50000*64 f16 shadow of x

    // Weight transpose/convert (once per launch)
    k_prep<<<(3*COUT*KPAD + AF*EPAD + CPAD*AF + 255)/256, 256, 0, stream>>>(
        fcw, emb_w, af_w, wT, ebT, afT);

    // Embedding GEMM
    k_embed<<<dim3((NN / 16 + 7) / 8), dim3(256), 0, stream>>>(atom_fea, ebT, xA, xh);

    float* bufs[2] = {xA, xB};
    const int convBlocks = (EDGES / 16 + 7) / 8;
    for (int l = 0; l < 3; ++l) {
        float* xin  = bufs[l & 1];
        float* xout = bufs[(l + 1) & 1];
        hipMemsetAsync(stats1, 0, (2 * COUT + 2 * AF) * sizeof(float), stream);
        hipMemsetAsync(summed, 0, (size_t)NN * AF * sizeof(float), stream);
        const _Float16* wl = wT + l * COUT * KPAD;
        const float* b = fcb + l * COUT;
        k_conv_stats<<<convBlocks, 256, 0, stream>>>(xh, nbr_fea, nbr_idx, wl, b, stats1);
        k_conv_apply<<<convBlocks, 256, 0, stream>>>(xh, nbr_fea, nbr_idx, wl, b, stats1,
                                                     bn1g + l * COUT, bn1b + l * COUT, summed);
        k_bn2_stats<<<128, 256, 0, stream>>>(summed, stats2);
        k_bn2_apply<<<512, 256, 0, stream>>>(xin, summed, stats2,
                                             bn2g + l * AF, bn2b + l * AF, xout, xh);
    }
    float* xf = bufs[1];   // after layers 0->1, 1->0, 0->1

    k_pool<<<(BB * NPC + 31) / 32, 192, 0, stream>>>(xf, cidx, adj_w, adj_b,
                                                     fc1_w, fc1_b, logp);
    k_edge_prob<<<2048, 256, 0, stream>>>(logp, out);
    k_atom_feat<<<((BB * NPC) / 16 + 7) / 8, 256, 0, stream>>>(
        xh, cidx, afT, af_b, out + (long)BB * NPC * NPC * 6);
}
